// TitansEEGMAC_57578331570295
// MI455X (gfx1250) — compile-verified, matrix operand order checked by reading, not run on hardware
//
#include <hip/hip_runtime.h>
#include <hip/hip_bf16.h>

// ---------------- problem constants ----------------
constexpr int Bb   = 32;
constexpr int Cc   = 64;
constexpr int Tt   = 4096;
constexpr int Dd   = 256;
constexpr int SEGc = 128;
constexpr int Sc   = 32;     // T / SEG
constexpr int NLc  = 16;
constexpr int Nc   = SEGc + NLc;   // 144
constexpr int NPc  = 4;
constexpr int Hc   = 4;
constexpr int DHc  = 64;
constexpr int FFc  = 1024;
constexpr int DEPTHc = 2;
constexpr int KFc  = NPc + Nc;     // 148 kv rows
constexpr int KFp  = 160;          // padded to 16
constexpr long long BT = (long long)Bb * Tt;          // 131072 rows
constexpr long long MA = (long long)Bb * Sc * Nc;     // 147456 rows

// ---------------- WMMA types ----------------
typedef __attribute__((ext_vector_type(16))) __bf16 v16bf;
typedef __attribute__((ext_vector_type(8)))  float  v8f;

union FragU { v16bf v; uint4 q[2]; };

__device__ inline v8f wmma_bf16(v16bf a, v16bf b, v8f c) {
  return __builtin_amdgcn_wmma_f32_16x16x32_bf16(false, a, false, b, (short)0, c, false, false);
}

// A/B fragment (16x32 bf16) from row-major bf16 (LDS or global).
// Lane layout per CDNA5 ISA: lanes 0-15: K = {0..7,16..23}; lanes 16-31: K = {8..15,24..31}.
__device__ inline v16bf ld_frag_bf16(const __bf16* p0, int ld, int row0, int k0) {
  const int lane = (int)(threadIdx.x & 31);
  const int r = lane & 15, hi = lane >> 4;
  const __bf16* p = p0 + (size_t)(row0 + r) * ld + k0 + hi * 8;
  FragU f;
  f.q[0] = *(const uint4*)(p);
  f.q[1] = *(const uint4*)(p + 16);
  return f.v;
}

// Same fragment, sourced from row-major f32 in global, converted to bf16 on the fly.
__device__ inline v16bf ld_frag_f32(const float* p0, int ld, int row0, int k0) {
  const int lane = (int)(threadIdx.x & 31);
  const int r = lane & 15, hi = lane >> 4;
  const float* p = p0 + (size_t)(row0 + r) * ld + k0 + hi * 8;
  float4 a0 = *(const float4*)(p);
  float4 a1 = *(const float4*)(p + 4);
  float4 b0 = *(const float4*)(p + 16);
  float4 b1 = *(const float4*)(p + 20);
  v16bf v;
  v[0]=(__bf16)a0.x; v[1]=(__bf16)a0.y; v[2]=(__bf16)a0.z; v[3]=(__bf16)a0.w;
  v[4]=(__bf16)a1.x; v[5]=(__bf16)a1.y; v[6]=(__bf16)a1.z; v[7]=(__bf16)a1.w;
  v[8]=(__bf16)b0.x; v[9]=(__bf16)b0.y; v[10]=(__bf16)b0.z; v[11]=(__bf16)b0.w;
  v[12]=(__bf16)b1.x; v[13]=(__bf16)b1.y; v[14]=(__bf16)b1.z; v[15]=(__bf16)b1.w;
  return v;
}

__device__ inline float gelu_f(float x) {
  float x3 = x * x * x;
  return 0.5f * x * (1.0f + tanhf(0.7978845608028654f * (x + 0.044715f * x3)));
}

// CDNA5 async global->LDS 16-byte copy per lane (ASYNCcnt-tracked), with portable fallback.
__device__ inline void async_copy_b128(void* ldsDst, const void* gSrc) {
#if defined(__gfx1250__)
  unsigned ldsOff = (unsigned)(uintptr_t)ldsDst;   // LDS aperture: low 32 bits = LDS byte offset
  asm volatile("global_load_async_to_lds_b128 %0, %1, off"
               :: "v"(ldsOff), "v"(gSrc)
               : "memory");
#else
  *(uint4*)ldsDst = *(const uint4*)gSrc;
#endif
}

__device__ inline void async_wait0() {
#if defined(__gfx1250__)
  asm volatile("s_wait_asynccnt 0" ::: "memory");
#endif
}

// ---------------- small data-movement kernels ----------------

// x[b,c,t] -> xt[(b*T+t), c]
__global__ void k_trx(const float* __restrict__ x, float* __restrict__ xt) {
  size_t idx = (size_t)blockIdx.x * 256 + threadIdx.x;
  if (idx >= (size_t)Bb * Cc * Tt) return;
  int c = (int)(idx & 63);
  size_t r = idx >> 6;           // b*T + t
  int t = (int)(r & 4095);
  size_t b = r >> 12;
  xt[idx] = x[(b * Cc + c) * Tt + t];
}

// Pack W[K x N] (f32 row-major) -> Wt[N x K] bf16 (B-operand layout)
__global__ void k_pack(const float* __restrict__ W, __bf16* __restrict__ Wt, int K, int Nn) {
  int idx = blockIdx.x * 256 + threadIdx.x;
  if (idx >= K * Nn) return;
  int k = idx % K, n = idx / K;
  Wt[idx] = (__bf16)W[(size_t)k * Nn + n];
}

// Mt[b][e][d] = M0[d][e] for all b
__global__ void k_initM(const float* __restrict__ M0, float* __restrict__ Mt) {
  int idx = blockIdx.x * 256 + threadIdx.x;
  if (idx >= Bb * Dd * Dd) return;
  int d = idx & 255;
  int e = (idx >> 8) & 255;
  Mt[idx] = M0[(size_t)d * Dd + e];
}

// a[b,s,SEG+j,:] = mem_tokens[j,:]
__global__ void k_fillmem(const float* __restrict__ mt, float* __restrict__ a) {
  size_t idx = (size_t)blockIdx.x * 256 + threadIdx.x;
  if (idx >= (size_t)Bb * Sc * NLc * Dd) return;
  int d = (int)(idx & 255);
  size_t r = idx >> 8;
  int j = (int)(r % NLc);
  size_t bs = r / NLc;
  a[(bs * Nc + SEGc + j) * Dd + d] = mt[(size_t)j * Dd + d];
}

// h[b, s*SEG+t, :] = a[b,s,t,:] (t < SEG)
__global__ void k_slice(const float* __restrict__ a, float* __restrict__ h) {
  size_t idx = (size_t)blockIdx.x * 256 + threadIdx.x;
  if (idx >= (size_t)Bb * Tt * Dd) return;
  int d = (int)(idx & 255);
  size_t r = idx >> 8;           // b*T + t
  size_t b = r >> 12;
  int tt = (int)(r & 4095);
  int s = tt >> 7, t = tt & 127;
  h[idx] = a[(((b * Sc + s) * Nc) + t) * Dd + d];
}

// ---------------- LayerNorm (D = 256), one wave per row ----------------
__global__ void __launch_bounds__(256) k_ln(const float* __restrict__ x, float* __restrict__ y,
                                            const float* __restrict__ sc, const float* __restrict__ bi,
                                            int M) {
  int wave = threadIdx.x >> 5, lane = threadIdx.x & 31;
  int row = blockIdx.x * 8 + wave;
  if (row >= M) return;
  const float* xr = x + (size_t)row * Dd;
  float v[8]; float s = 0.f;
#pragma unroll
  for (int j = 0; j < 8; ++j) { v[j] = xr[lane + 32 * j]; s += v[j]; }
#pragma unroll
  for (int o = 16; o > 0; o >>= 1) s += __shfl_xor(s, o, 32);
  float mean = s * (1.0f / 256.0f);
  float q = 0.f;
#pragma unroll
  for (int j = 0; j < 8; ++j) { float d = v[j] - mean; q += d * d; }
#pragma unroll
  for (int o = 16; o > 0; o >>= 1) q += __shfl_xor(q, o, 32);
  float inv = rsqrtf(q * (1.0f / 256.0f) + 1e-5f);
  float* yr = y + (size_t)row * Dd;
#pragma unroll
  for (int j = 0; j < 8; ++j) {
    int c = lane + 32 * j;
    yr[c] = (v[j] - mean) * inv * sc[c] + bi[c];
  }
}

// ---------------- generic tiled WMMA GEMM ----------------
// C[M x Nn] = epi(A[M x K] @ Wt^T + ...) ; Wt is bf16 [Nn x K] (B-operand layout).
// Block: 256 threads = 8 waves (2 M-tiles x 4 N-tiles), output 32x64 per block.
// B tile is staged global->LDS with CDNA5 async-to-LDS ops; A tile converts f32->bf16 in VGPRs.
// EPI: 0 store, 1 +bias, 2 +bias,gelu, 3 +bias,+res, 4 C += result
template <typename AT, typename OT, int EPI>
__global__ void __launch_bounds__(256) k_gemm(const AT* __restrict__ A, const __bf16* __restrict__ Wt,
                                              const float* __restrict__ bias, const float* __restrict__ res,
                                              OT* __restrict__ C, int M, int K, int Nn) {
  const int m0 = blockIdx.x * 32;
  const int n0 = blockIdx.y * 64;
  const int tid = threadIdx.x;
  const int wave = tid >> 5, lane = tid & 31;
  const int wm = wave >> 2, wn = wave & 3;
  __shared__ __bf16 sA[32 * 40];
  __shared__ __bf16 sB[64 * 40];
  v8f acc = {};
  const int rA = tid >> 3, cA = (tid & 7) * 4;       // A stage coords
  const int rB = tid >> 2, cB = (tid & 3) * 8;       // B stage coords (16B chunks)
  for (int k0 = 0; k0 < K; k0 += 32) {
    __syncthreads();
    // stage B tile 64x32 via async global->LDS (ASYNCcnt)
    async_copy_b128(&sB[rB * 40 + cB], Wt + (size_t)(n0 + rB) * K + k0 + cB);
    // stage A tile 32x32 (convert -> bf16), prefetch next K tile
    {
      const AT* ap = A + (size_t)(m0 + rA) * K + k0 + cA;
#pragma unroll
      for (int j = 0; j < 4; ++j) sA[rA * 40 + cA + j] = (__bf16)(float)ap[j];
      if (k0 + 32 < K) __builtin_prefetch(ap + 32, 0, 1);
    }
    async_wait0();
    __syncthreads();
    v16bf af = ld_frag_bf16(sA, 40, wm * 16, 0);
    v16bf bf = ld_frag_bf16(sB, 40, wn * 16, 0);
    acc = wmma_bf16(af, bf, acc);
  }
  const int hi = lane >> 4, nn = lane & 15;
#pragma unroll
  for (int i = 0; i < 8; ++i) {
    int row = m0 + wm * 16 + i + 8 * hi;
    int col = n0 + wn * 16 + nn;
    size_t off = (size_t)row * Nn + col;
    float v = acc[i];
    if (EPI == 1 || EPI == 2 || EPI == 3) v += bias[col];
    if (EPI == 2) v = gelu_f(v);
    if (EPI == 3) v += res[off];
    if (EPI == 4) v += (float)C[off];
    C[off] = (OT)v;
  }
}

// ---------------- memory-matrix scan ----------------
// One workgroup per batch. LDS (dynamic, 256 KB): qL[128x256], kL[128x256],
// kT[256x128], eT[256x128] bf16. State Mt[b] = M^T in global f32, updated per segment.
__global__ void __launch_bounds__(256) k_scan(const float* __restrict__ h,
                                              const __bf16* __restrict__ WmqP,
                                              const __bf16* __restrict__ WmkP,
                                              const __bf16* __restrict__ WmvP,
                                              float* __restrict__ Mt,
                                              float* __restrict__ a) {
  extern __shared__ char smem[];
  __bf16* qL = (__bf16*)smem;             // 128*256
  __bf16* kL = qL + 128 * 256;            // 128*256
  __bf16* kT = kL + 128 * 256;            // 256*128
  __bf16* eT = kT + 256 * 128;            // 256*128
  const int b = blockIdx.x;
  const int tid = threadIdx.x, wave = tid >> 5, lane = tid & 31;
  const int hi = lane >> 4, nn = lane & 15;
  float* Mb = Mt + (size_t)b * Dd * Dd;
  const float c_lr = 0.1f / (float)SEGc;
  for (int s = 0; s < Sc; ++s) {
    const float* X = h + ((size_t)b * Tt + (size_t)s * SEGc) * Dd;  // [128 x 256]
    // phase 1: q = X@Wmq, k = X@Wmk  (store q, k, k^T in LDS as bf16)
    for (int t = wave; t < 128; t += 8) {
      int tm = t & 7, tn = t >> 3;
      v8f aq = {}, ak = {};
      for (int k0 = 0; k0 < Dd; k0 += 32) {
        v16bf xa = ld_frag_f32(X, Dd, tm * 16, k0);
        aq = wmma_bf16(xa, ld_frag_bf16(WmqP, Dd, tn * 16, k0), aq);
        ak = wmma_bf16(xa, ld_frag_bf16(WmkP, Dd, tn * 16, k0), ak);
      }
#pragma unroll
      for (int i = 0; i < 8; ++i) {
        int row = tm * 16 + i + 8 * hi, col = tn * 16 + nn;
        qL[row * 256 + col] = (__bf16)aq[i];
        __bf16 kv = (__bf16)ak[i];
        kL[row * 256 + col] = kv;
        kT[col * 128 + row] = kv;
      }
    }
    __syncthreads();
    // phase 2: err^T = (X@Wmv - k@M)^T ; r = q@M ; a[seg rows] = X + r
    for (int t = wave; t < 128; t += 8) {
      int tm = t & 7, tn = t >> 3;
      v8f av = {}, akM = {}, aqM = {};
      for (int k0 = 0; k0 < Dd; k0 += 32) {
        v16bf xa = ld_frag_f32(X, Dd, tm * 16, k0);
        av = wmma_bf16(xa, ld_frag_bf16(WmvP, Dd, tn * 16, k0), av);
        v16bf bM = ld_frag_f32(Mb, Dd, tn * 16, k0);  // Mt rows are output cols
        akM = wmma_bf16(ld_frag_bf16(kL, 256, tm * 16, k0), bM, akM);
        aqM = wmma_bf16(ld_frag_bf16(qL, 256, tm * 16, k0), bM, aqM);
      }
#pragma unroll
      for (int i = 0; i < 8; ++i) {
        int row = tm * 16 + i + 8 * hi, col = tn * 16 + nn;
        eT[col * 128 + row] = (__bf16)(av[i] - akM[i]);
        size_t aoff = (((size_t)(b * Sc + s) * Nc) + row) * Dd + col;
        a[aoff] = X[(size_t)row * Dd + col] + aqM[i];
      }
    }
    __syncthreads();
    // phase 3: Mt[e][d] += c * sum_s errT[e][s] * kT[d][s]
    for (int t = wave; t < 256; t += 8) {
      int te = t & 15, td = t >> 4;
      v8f u = {};
      for (int k0 = 0; k0 < SEGc; k0 += 32) {
        u = wmma_bf16(ld_frag_bf16(eT, 128, te * 16, k0),
                      ld_frag_bf16(kT, 128, td * 16, k0), u);
      }
#pragma unroll
      for (int i = 0; i < 8; ++i) {
        int e = te * 16 + i + 8 * hi, d = td * 16 + nn;
        size_t off = (size_t)e * Dd + d;
        Mb[off] += c_lr * u[i];
      }
    }
    __threadfence();
    __syncthreads();
  }
}

// ---------------- attention per (b,s,head) ----------------
// LDS (dynamic ~193 KB): qS[144x64], kS[160x64], vT[64x160] bf16; sS[144x160] f32; pS[144x160] bf16.
__global__ void __launch_bounds__(256) k_attn(const float* __restrict__ qb, const float* __restrict__ kb,
                                              const float* __restrict__ vb, const float* __restrict__ pk,
                                              const float* __restrict__ pv, float* __restrict__ ob) {
  extern __shared__ char smem[];
  __bf16* qS = (__bf16*)smem;                     // 144*64
  __bf16* kS = qS + Nc * DHc;                     // 160*64
  __bf16* vT = kS + KFp * DHc;                    // 64*160
  float*  sS = (float*)(vT + DHc * KFp);          // 144*160
  __bf16* pS = (__bf16*)(sS + Nc * KFp);          // 144*160
  const int blk = blockIdx.x;
  const int hh = blk & (Hc - 1);
  const int bs = blk >> 2;
  const int tid = threadIdx.x;
  const size_t baseRow = (size_t)bs * Nc;
  for (int idx = tid; idx < Nc * DHc; idx += 256) {
    int n = idx >> 6, dh = idx & 63;
    qS[idx] = (__bf16)qb[(baseRow + n) * Dd + hh * DHc + dh];
  }
  for (int idx = tid; idx < KFp * DHc; idx += 256) {
    int j = idx >> 6, dh = idx & 63;
    float kv = 0.f, vv = 0.f;
    if (j < NPc) {
      kv = pk[((size_t)hh * NPc + j) * DHc + dh];
      vv = pv[((size_t)hh * NPc + j) * DHc + dh];
    } else if (j < KFc) {
      kv = kb[(baseRow + (j - NPc)) * Dd + hh * DHc + dh];
      vv = vb[(baseRow + (j - NPc)) * Dd + hh * DHc + dh];
    }
    kS[j * DHc + dh] = (__bf16)kv;
    vT[dh * KFp + j] = (__bf16)vv;
  }
  __syncthreads();
  const int wave = tid >> 5, lane = tid & 31, hi = lane >> 4, nn = lane & 15;
  // scores = q @ kf^T / 8, masked
  for (int t = wave; t < 9 * 10; t += 8) {
    int tm = t / 10, tn = t % 10;
    v8f acc = {};
    for (int k0 = 0; k0 < DHc; k0 += 32)
      acc = wmma_bf16(ld_frag_bf16(qS, DHc, tm * 16, k0),
                      ld_frag_bf16(kS, DHc, tn * 16, k0), acc);
#pragma unroll
    for (int i = 0; i < 8; ++i) {
      int row = tm * 16 + i + 8 * hi, col = tn * 16 + nn;
      float v = acc[i] * 0.125f;  // 1/sqrt(64)
      bool ok = (col < KFc) && (col < row + NPc + 1);
      sS[row * KFp + col] = ok ? v : -1e9f;
    }
  }
  __syncthreads();
  // per-row softmax
  if (tid < Nc) {
    float m = -1e30f;
    for (int c = 0; c < KFc; ++c) m = fmaxf(m, sS[tid * KFp + c]);
    float sum = 0.f;
    for (int c = 0; c < KFp; ++c) {
      float e = (c < KFc) ? expf(sS[tid * KFp + c] - m) : 0.f;
      sS[tid * KFp + c] = e;
      sum += e;
    }
    float inv = 1.0f / sum;
    for (int c = 0; c < KFp; ++c) pS[tid * KFp + c] = (__bf16)(sS[tid * KFp + c] * inv);
  }
  __syncthreads();
  // o = P @ vf
  for (int t = wave; t < 9 * 4; t += 8) {
    int tm = t >> 2, tn = t & 3;
    v8f acc = {};
    for (int k0 = 0; k0 < KFp; k0 += 32)
      acc = wmma_bf16(ld_frag_bf16(pS, KFp, tm * 16, k0),
                      ld_frag_bf16(vT, KFp, tn * 16, k0), acc);
#pragma unroll
    for (int i = 0; i < 8; ++i) {
      int row = tm * 16 + i + 8 * hi, col = tn * 16 + nn;
      ob[(baseRow + row) * Dd + hh * DHc + col] = acc[i];
    }
  }
}

// ---------------- final LN + deterministic mean pooling ----------------
// block (b, chunk of 128 t-rows): partial[b][chunk][d] = sum over 128 rows of lnf(h)[d]
__global__ void __launch_bounds__(256) k_lnmean(const float* __restrict__ h, const float* __restrict__ sc,
                                                const float* __restrict__ bi, float* __restrict__ partial) {
  __shared__ float red[8 * 256];
  int b = blockIdx.x, chunk = blockIdx.y;
  int wave = threadIdx.x >> 5, lane = threadIdx.x & 31;
  float pacc[8];
#pragma unroll
  for (int j = 0; j < 8; ++j) pacc[j] = 0.f;
  for (int rr = 0; rr < 16; ++rr) {
    int trow = chunk * 128 + wave * 16 + rr;
    const float* xr = h + ((size_t)b * Tt + trow) * Dd;
    float v[8]; float s = 0.f;
#pragma unroll
    for (int j = 0; j < 8; ++j) { v[j] = xr[lane + 32 * j]; s += v[j]; }
#pragma unroll
    for (int o = 16; o > 0; o >>= 1) s += __shfl_xor(s, o, 32);
    float mean = s * (1.0f / 256.0f);
    float q = 0.f;
#pragma unroll
    for (int j = 0; j < 8; ++j) { float d = v[j] - mean; q += d * d; }
#pragma unroll
    for (int o = 16; o > 0; o >>= 1) q += __shfl_xor(q, o, 32);
    float inv = rsqrtf(q * (1.0f / 256.0f) + 1e-5f);
#pragma unroll
    for (int j = 0; j < 8; ++j) {
      int c = lane + 32 * j;
      pacc[j] += (v[j] - mean) * inv * sc[c] + bi[c];
    }
  }
#pragma unroll
  for (int j = 0; j < 8; ++j) red[wave * 256 + lane + 32 * j] = pacc[j];
  __syncthreads();
  int t = threadIdx.x;
  float sum = 0.f;
#pragma unroll
  for (int w = 0; w < 8; ++w) sum += red[w * 256 + t];
  partial[((size_t)b * 32 + chunk) * 256 + t] = sum;
}

__global__ void k_meanfin(const float* __restrict__ partial, float* __restrict__ meanln) {
  int idx = blockIdx.x * 256 + threadIdx.x;
  if (idx >= Bb * Dd) return;
  int b = idx >> 8, d = idx & 255;
  float s = 0.f;
  for (int c = 0; c < 32; ++c) s += partial[((size_t)b * 32 + c) * 256 + d];
  meanln[idx] = s * (1.0f / (float)Tt);
}

// pooled = meanln @ W_logit + b_logit ; out = pooled @ W_cls + b_cls
__global__ void __launch_bounds__(256) k_head(const float* __restrict__ meanln,
                                              const float* __restrict__ Wl, const float* __restrict__ bl,
                                              const float* __restrict__ Wc, const float* __restrict__ bc,
                                              float* __restrict__ out) {
  __shared__ float pooled[32 * 256];
  int tid = threadIdx.x;
  for (int idx = tid; idx < 32 * 256; idx += 256) {
    int b = idx >> 8, e = idx & 255;
    float s = bl[e];
    for (int d = 0; d < 256; ++d) s += meanln[b * 256 + d] * Wl[(size_t)d * 256 + e];
    pooled[idx] = s;
  }
  __syncthreads();
  if (tid < 32 * 4) {
    int b = tid >> 2, c = tid & 3;
    float s = bc[c];
    for (int e = 0; e < 256; ++e) s += pooled[b * 256 + e] * Wc[e * 4 + c];
    out[b * 4 + c] = s;
  }
}

// ---------------- host orchestration ----------------
extern "C" void kernel_launch(void* const* d_in, const int* in_sizes, int n_in,
                              void* d_out, int out_size, void* d_ws, size_t ws_size,
                              hipStream_t stream) {
  (void)in_sizes; (void)n_in; (void)out_size; (void)ws_size;
  const float* x        = (const float*)d_in[0];
  const float* W_in     = (const float*)d_in[1];
  const float* b_in     = (const float*)d_in[2];
  const float* mem_tok  = (const float*)d_in[3];
  const float* pk       = (const float*)d_in[4];
  const float* pv       = (const float*)d_in[5];
  const float* Wq       = (const float*)d_in[6];
  const float* Wk       = (const float*)d_in[7];
  const float* Wv       = (const float*)d_in[8];
  const float* Wo       = (const float*)d_in[9];
  const float* ln1_s    = (const float*)d_in[10];
  const float* ln1_b    = (const float*)d_in[11];
  const float* ln2_s    = (const float*)d_in[12];
  const float* ln2_b    = (const float*)d_in[13];
  const float* W1       = (const float*)d_in[14];
  const float* b1       = (const float*)d_in[15];
  const float* W2       = (const float*)d_in[16];
  const float* b2       = (const float*)d_in[17];
  const float* M0       = (const float*)d_in[18];
  const float* Wmq      = (const float*)d_in[19];
  const float* Wmk      = (const float*)d_in[20];
  const float* Wmv      = (const float*)d_in[21];
  const float* lnf_s    = (const float*)d_in[22];
  const float* lnf_b    = (const float*)d_in[23];
  const float* W_logit  = (const float*)d_in[24];
  const float* b_logit  = (const float*)d_in[25];
  const float* W_cls    = (const float*)d_in[26];
  const float* b_cls    = (const float*)d_in[27];
  float* out = (float*)d_out;

  // ---- workspace carve-up ----
  char* ws = (char*)d_ws;
  size_t off = 0;
  auto take = [&](size_t bytes) -> char* {
    char* p = ws + off;
    off += (bytes + 255) & ~(size_t)255;
    return p;
  };
  float*  xt      = (float*)take((size_t)BT * Cc * 4);
  float*  h       = (float*)take((size_t)BT * Dd * 4);
  float*  a       = (float*)take((size_t)MA * Dd * 4);
  float*  u       = (float*)take((size_t)MA * Dd * 4);   // ln1 out; reused as attn output (pre-Wo)
  float*  qb      = (float*)take((size_t)MA * Dd * 4);   // q; reused as ln2 out (f)
  float*  kb      = (float*)take((size_t)MA * Dd * 4);
  float*  vb      = (float*)take((size_t)MA * Dd * 4);
  __bf16* g       = (__bf16*)take((size_t)MA * FFc * 2); // gelu(f@W1+b1)
  float*  Mt      = (float*)take((size_t)Bb * Dd * Dd * 4);
  float*  partial = (float*)take((size_t)Bb * 32 * Dd * 4);
  float*  meanln  = (float*)take((size_t)Bb * Dd * 4);
  __bf16* WinP    = (__bf16*)take((size_t)Dd * Cc * 2);
  const size_t DD = (size_t)Dd * Dd;
  const size_t DF = (size_t)Dd * FFc;
  __bf16* packL[DEPTHc];
  for (int l = 0; l < DEPTHc; ++l)
    packL[l] = (__bf16*)take((7 * DD + 2 * DF) * 2);

  auto pack = [&](const float* W, __bf16* Wt, int K, int Nn) {
    int tot = K * Nn;
    hipLaunchKernelGGL(k_pack, dim3((tot + 255) / 256), dim3(256), 0, stream, W, Wt, K, Nn);
  };

  // ---- transpose x, pack all weights ----
  {
    size_t tot = (size_t)Bb * Cc * Tt;
    hipLaunchKernelGGL(k_trx, dim3((unsigned)((tot + 255) / 256)), dim3(256), 0, stream, x, xt);
  }
  pack(W_in, WinP, Cc, Dd);
  for (int l = 0; l < DEPTHc; ++l) {
    __bf16* P = packL[l];
    pack(Wmq + l * DD, P + 0 * DD, Dd, Dd);
    pack(Wmk + l * DD, P + 1 * DD, Dd, Dd);
    pack(Wmv + l * DD, P + 2 * DD, Dd, Dd);
    pack(Wq  + l * DD, P + 3 * DD, Dd, Dd);
    pack(Wk  + l * DD, P + 4 * DD, Dd, Dd);
    pack(Wv  + l * DD, P + 5 * DD, Dd, Dd);
    pack(Wo  + l * DD, P + 6 * DD, Dd, Dd);
    pack(W1  + l * DF, P + 7 * DD, Dd, FFc);            // [FF x D]
    pack(W2  + l * DF, P + 7 * DD + DF, FFc, Dd);       // [D x FF]
  }

  // ---- input projection: h = xt @ W_in + b_in ----
  hipLaunchKernelGGL((k_gemm<float, float, 1>), dim3((unsigned)(BT / 32), Dd / 64), dim3(256), 0, stream,
                     xt, WinP, b_in, nullptr, h, (int)BT, Cc, Dd);

  const size_t scanLds = 4u * 128 * 256 * sizeof(__bf16);          // 256 KB
  const size_t attnLds = (size_t)Nc * DHc * 2 + (size_t)KFp * DHc * 2 + (size_t)DHc * KFp * 2
                       + (size_t)Nc * KFp * 4 + (size_t)Nc * KFp * 2; // ~193 KB

  for (int l = 0; l < DEPTHc; ++l) {
    __bf16* P = packL[l];
    // memory scan: a[seg rows] = h + r ; Mt updated sequentially (one WG per batch, 256KB LDS)
    hipLaunchKernelGGL(k_initM, dim3((Bb * Dd * Dd) / 256), dim3(256), 0, stream, M0 + l * DD, Mt);
    hipLaunchKernelGGL(k_scan, dim3(Bb), dim3(256), scanLds, stream,
                       h, P + 0 * DD, P + 1 * DD, P + 2 * DD, Mt, a);
    hipLaunchKernelGGL(k_fillmem, dim3((unsigned)(((size_t)Bb * Sc * NLc * Dd) / 256)), dim3(256), 0, stream,
                       mem_tok + (size_t)l * NLc * Dd, a);
    // ln1 -> u
    hipLaunchKernelGGL(k_ln, dim3((unsigned)(MA / 8)), dim3(256), 0, stream,
                       a, u, ln1_s + l * Dd, ln1_b + l * Dd, (int)MA);
    // q,k,v projections
    hipLaunchKernelGGL((k_gemm<float, float, 0>), dim3((unsigned)(MA / 32), Dd / 64), dim3(256), 0, stream,
                       u, P + 3 * DD, nullptr, nullptr, qb, (int)MA, Dd, Dd);
    hipLaunchKernelGGL((k_gemm<float, float, 0>), dim3((unsigned)(MA / 32), Dd / 64), dim3(256), 0, stream,
                       u, P + 4 * DD, nullptr, nullptr, kb, (int)MA, Dd, Dd);
    hipLaunchKernelGGL((k_gemm<float, float, 0>), dim3((unsigned)(MA / 32), Dd / 64), dim3(256), 0, stream,
                       u, P + 5 * DD, nullptr, nullptr, vb, (int)MA, Dd, Dd);
    // attention (u becomes attention output pre-Wo)
    hipLaunchKernelGGL(k_attn, dim3(Bb * Sc * Hc), dim3(256), attnLds, stream,
                       qb, kb, vb, pk + (size_t)l * Hc * NPc * DHc, pv + (size_t)l * Hc * NPc * DHc, u);
    // a += o @ Wo
    hipLaunchKernelGGL((k_gemm<float, float, 4>), dim3((unsigned)(MA / 32), Dd / 64), dim3(256), 0, stream,
                       u, P + 6 * DD, nullptr, nullptr, a, (int)MA, Dd, Dd);
    // ln2 -> qb (reused as f)
    hipLaunchKernelGGL(k_ln, dim3((unsigned)(MA / 8)), dim3(256), 0, stream,
                       a, qb, ln2_s + l * Dd, ln2_b + l * Dd, (int)MA);
    // FF: g = gelu(f @ W1 + b1) as bf16 ; a = a + (g @ W2 + b2)
    hipLaunchKernelGGL((k_gemm<float, __bf16, 2>), dim3((unsigned)(MA / 32), FFc / 64), dim3(256), 0, stream,
                       qb, P + 7 * DD, b1 + l * FFc, nullptr, g, (int)MA, Dd, FFc);
    hipLaunchKernelGGL((k_gemm<__bf16, float, 3>), dim3((unsigned)(MA / 32), Dd / 64), dim3(256), 0, stream,
                       g, P + 7 * DD + DF, b2 + l * Dd, a, a, (int)MA, FFc, Dd);
    // h = a[:, :, :SEG, :]
    hipLaunchKernelGGL(k_slice, dim3((unsigned)(((size_t)Bb * Tt * Dd) / 256)), dim3(256), 0, stream, a, h);
  }

  // ---- final LN + mean pooling + head (linearity: mean commutes with W_logit) ----
  hipLaunchKernelGGL(k_lnmean, dim3(Bb, 32), dim3(256), 0, stream, h, lnf_s, lnf_b, partial);
  hipLaunchKernelGGL(k_meanfin, dim3((Bb * Dd) / 256), dim3(256), 0, stream, partial, meanln);
  hipLaunchKernelGGL(k_head, dim3(1), dim3(256), 0, stream, meanln, W_logit, b_logit, W_cls, b_cls, out);
}
